// ScaledDotProductAttn_85444079386690
// MI455X (gfx1250) — matrix operand order repaired and checked by reading, not compiled, not run on hardware
//
#include <hip/hip_runtime.h>

typedef __attribute__((ext_vector_type(16))) _Float16 v16h;
typedef __attribute__((ext_vector_type(8)))  _Float16 v8h;
typedef __attribute__((ext_vector_type(8)))  float    v8f;

#define S_LEN  2048
#define D_DIM  64
#define QT     64      // query rows per block (16 per wave x 4 waves)
#define KT     64      // keys per main-loop iteration
#define NIT    (S_LEN / KT)
#define LDST   72      // f16 LDS row stride (16B aligned, bank-rotating)
#define MST    68      // f32 mask LDS row stride (de-conflicts lane halves)
// fold (1/temperature) * log2(e) into the Q->f16 conversion; softmax done in exp2 domain
#define QSCALE (0.125f * 1.44269504088896f)

// ---- LDS layout (bytes). Ps aliases Qs: wave w reads Q rows [16w,16w+16) only
// for its resident aq fragments (before any P write), and writes P only in the
// same per-wave slice -> race-free alias. ----
#define OFF_QS  0                       // 64*72*2 = 9216   (Q tile, then P scratch)
#define OFF_KS  9216                    // 64*72*2 = 9216
#define OFF_VT  18432                   // 64*72*2 = 9216
#define OFF_MS  27648                   // 2 * 64*68*4 = 34816 (mask double buffer)
#define SMEM_SZ 62464

// ---- CDNA5 async global->LDS copy (ASYNCcnt path), with safe fallback ----
#if __has_builtin(__builtin_amdgcn_global_load_async_to_lds_b128)
typedef __attribute__((__vector_size__(4 * sizeof(int)))) int vi4;
typedef __attribute__((address_space(1))) vi4 gvi4;
typedef __attribute__((address_space(3))) vi4 lvi4;
__device__ __forceinline__ void async_copy16(const float* src, float* dst) {
  __builtin_amdgcn_global_load_async_to_lds_b128(
      (gvi4*)(void*)src, (lvi4*)(void*)dst, 0, 0);
}
// 8 copies per tile per thread: <=8 outstanding == newest tile only, so the
// previous tile's DMA is complete (ASYNCcnt is per-wave, in-order).
__device__ __forceinline__ void wait_async_prev_tile() {
#if __has_builtin(__builtin_amdgcn_s_wait_asynccnt)
  __builtin_amdgcn_s_wait_asynccnt(8);
#else
  asm volatile("s_wait_asynccnt 8" ::: "memory");
#endif
}
#else
__device__ __forceinline__ void async_copy16(const float* src, float* dst) {
  *(float4*)dst = *(const float4*)src;  // synchronous fallback
}
__device__ __forceinline__ void wait_async_prev_tile() {}
#endif

__device__ __forceinline__ float fast_exp2(float x) {
#if __has_builtin(__builtin_amdgcn_exp2f)
  return __builtin_amdgcn_exp2f(x);
#else
  return exp2f(x);
#endif
}

// A-fragment (16x32 f16): lane L holds row M=L%16; K-halves {hi*8+0..7, 16+hi*8+0..7}
__device__ __forceinline__ v16h load_fragA(const _Float16* base, int row, int kbase, int hi) {
  const _Float16* p = base + row * LDST + kbase + hi * 8;
  v8h lo = *(const v8h*)p;
  v8h hv = *(const v8h*)(p + 16);
  v16h r;
#pragma unroll
  for (int i = 0; i < 8; ++i) { r[i] = lo[i]; r[i + 8] = hv[i]; }
  return r;
}

// B-fragment (32x16 f16): lane L holds column N=L%16; K-halves {hi*8+0..7, 16+hi*8+0..7}
__device__ __forceinline__ v16h load_fragB(const _Float16* base, int col, int kbase, int hi) {
  const _Float16* p = base + col * LDST + kbase + hi * 8;
  v8h lo = *(const v8h*)p;
  v8h hv = *(const v8h*)(p + 16);
  v16h r;
#pragma unroll
  for (int i = 0; i < 8; ++i) { r[i] = lo[i]; r[i + 8] = hv[i]; }
  return r;
}

__global__ __launch_bounds__(128)
void fa_fwd_kernel(const float* __restrict__ q, const float* __restrict__ k,
                   const float* __restrict__ v, const float* __restrict__ mask,
                   float* __restrict__ out) {
  __shared__ __align__(16) char smem[SMEM_SZ];
  _Float16* Qs  = (_Float16*)(smem + OFF_QS);   // [qrow][d]; later per-wave P scratch
  _Float16* Ks  = (_Float16*)(smem + OFF_KS);   // [key][d]
  _Float16* VTs = (_Float16*)(smem + OFF_VT);   // [d][key]
  float*    Ms0 = (float*)(smem + OFF_MS);      // mask tile ping
  float*    Ms1 = Ms0 + QT * MST;               // mask tile pong

  const int tid  = threadIdx.x;
  const int wave = tid >> 5;
  const int lane = tid & 31;
  const int ln   = lane & 15;
  const int hi   = lane >> 4;

  const int bh = blockIdx.y;            // b*H + h
  const int q0 = blockIdx.x * QT;

  const float* qg = q + (size_t)bh * S_LEN * D_DIM;
  const float* kg = k + (size_t)bh * S_LEN * D_DIM;
  const float* vg = v + (size_t)bh * S_LEN * D_DIM;

  const int srow = tid >> 1;            // staging row (0..63), 2 threads/row
  const int sdh  = (tid & 1) * 32;      // staging d-half

  // ---- stage Q tile: 64x64 f32 -> f16, pre-scaled by (1/T)*log2(e) ----
  {
    const float4* g = (const float4*)(qg + (size_t)(q0 + srow) * D_DIM + sdh);
#pragma unroll
    for (int i = 0; i < 8; ++i) {
      float4 f = g[i];
      _Float16* d = &Qs[srow * LDST + sdh + i * 4];
      d[0] = (_Float16)(f.x * QSCALE); d[1] = (_Float16)(f.y * QSCALE);
      d[2] = (_Float16)(f.z * QSCALE); d[3] = (_Float16)(f.w * QSCALE);
    }
  }

  // ---- preload K/V tile 0 into registers; start mask tile 0 DMA ----
  float4 kreg[8], vreg[8];
  {
    const float4* gk = (const float4*)(kg + (size_t)srow * D_DIM + sdh);
    const float4* gv = (const float4*)(vg + (size_t)srow * D_DIM + sdh);
#pragma unroll
    for (int i = 0; i < 8; ++i) { kreg[i] = gk[i]; vreg[i] = gv[i]; }
  }
#pragma unroll
  for (int i = 0; i < 8; ++i)
    async_copy16(mask + (size_t)(q0 + srow) * S_LEN + sdh + i * 4,
                 &Ms0[srow * MST + sdh + i * 4]);
  __syncthreads();

  // per-wave Q A-fragments, resident across the whole key loop (alias-safe: own slice)
  const v16h aq0 = load_fragA(Qs, wave * 16 + ln, 0, hi);
  const v16h aq1 = load_fragA(Qs, wave * 16 + ln, 32, hi);
  _Float16* Pw = Qs + wave * 16 * LDST;   // P scratch aliases this wave's Q slice

  float mrun[8], lrun[8];
  v8f o[4];
#pragma unroll
  for (int j = 0; j < 8; ++j) { mrun[j] = -INFINITY; lrun[j] = 0.0f; }
#pragma unroll
  for (int n = 0; n < 4; ++n)
#pragma unroll
    for (int j = 0; j < 8; ++j) o[n][j] = 0.0f;

  for (int it = 0; it < NIT; ++it) {
    const int kb = it * KT;
    const float* Mcur = (it & 1) ? Ms1 : Ms0;
    float*       Mnxt = (it & 1) ? Ms0 : Ms1;

    // ---- drain K/V registers -> LDS (cvt to f16; V transposed) ----
#pragma unroll
    for (int i = 0; i < 8; ++i) {
      float4 f = kreg[i];
      _Float16* d = &Ks[srow * LDST + sdh + i * 4];
      d[0] = (_Float16)f.x; d[1] = (_Float16)f.y;
      d[2] = (_Float16)f.z; d[3] = (_Float16)f.w;
    }
#pragma unroll
    for (int i = 0; i < 8; ++i) {
      float4 f = vreg[i];
      VTs[(sdh + i * 4 + 0) * LDST + srow] = (_Float16)f.x;
      VTs[(sdh + i * 4 + 1) * LDST + srow] = (_Float16)f.y;
      VTs[(sdh + i * 4 + 2) * LDST + srow] = (_Float16)f.z;
      VTs[(sdh + i * 4 + 3) * LDST + srow] = (_Float16)f.w;
    }

    // ---- issue NEXT mask tile DMA; wait for PREVIOUS tile's DMA ----
    {
      const int kbn = (kb + KT) & (S_LEN - 1);   // wrap: keeps wait pattern uniform
#pragma unroll
      for (int i = 0; i < 8; ++i)
        async_copy16(mask + (size_t)(q0 + srow) * S_LEN + kbn + sdh + i * 4,
                     &Mnxt[srow * MST + sdh + i * 4]);
    }
    wait_async_prev_tile();
    __syncthreads();

    // ---- issue next K/V tile global loads (latency hidden under compute) ----
    if (kb + KT < S_LEN) {
      const float4* gk = (const float4*)(kg + (size_t)(kb + KT + srow) * D_DIM + sdh);
      const float4* gv = (const float4*)(vg + (size_t)(kb + KT + srow) * D_DIM + sdh);
#pragma unroll
      for (int i = 0; i < 8; ++i) { kreg[i] = gk[i]; vreg[i] = gv[i]; }
    }

    // ---- T = (Q*scale) K^T : 16x64 tile per wave (8 WMMA), exp2-domain scores ----
    v8f s[4];
#pragma unroll
    for (int n = 0; n < 4; ++n)
#pragma unroll
      for (int j = 0; j < 8; ++j) s[n][j] = 0.0f;
#pragma unroll
    for (int n = 0; n < 4; ++n) {
      v16h bk0 = load_fragB(Ks, n * 16 + ln, 0, hi);
      v16h bk1 = load_fragB(Ks, n * 16 + ln, 32, hi);
      s[n] = __builtin_amdgcn_wmma_f32_16x16x32_f16(false, aq0, false, bk0,
                                                    (short)0, s[n], false, false);
      s[n] = __builtin_amdgcn_wmma_f32_16x16x32_f16(false, aq1, false, bk1,
                                                    (short)0, s[n], false, false);
    }

    // ---- multiplicative mask (from LDS tile); scale already folded into Q ----
#pragma unroll
    for (int n = 0; n < 4; ++n) {
      const int cl = n * 16 + ln;
#pragma unroll
      for (int j = 0; j < 8; ++j) {
        const int rl = wave * 16 + j + hi * 8;   // D-fragment: M = j + hi*8
        s[n][j] *= Mcur[rl * MST + cl];
      }
    }

    // ---- online softmax in exp2 domain: row reduce across 16-lane halves ----
#pragma unroll
    for (int j = 0; j < 8; ++j) {
      float rm = fmaxf(fmaxf(s[0][j], s[1][j]), fmaxf(s[2][j], s[3][j]));
      rm = fmaxf(rm, __shfl_xor(rm, 1));
      rm = fmaxf(rm, __shfl_xor(rm, 2));
      rm = fmaxf(rm, __shfl_xor(rm, 4));
      rm = fmaxf(rm, __shfl_xor(rm, 8));
      const float mnew  = fmaxf(mrun[j], rm);
      const float alpha = fast_exp2(mrun[j] - mnew);
      float p0 = fast_exp2(s[0][j] - mnew);
      float p1 = fast_exp2(s[1][j] - mnew);
      float p2 = fast_exp2(s[2][j] - mnew);
      float p3 = fast_exp2(s[3][j] - mnew);
      s[0][j] = p0; s[1][j] = p1; s[2][j] = p2; s[3][j] = p3;
      float rs = p0 + p1 + p2 + p3;
      rs += __shfl_xor(rs, 1);
      rs += __shfl_xor(rs, 2);
      rs += __shfl_xor(rs, 4);
      rs += __shfl_xor(rs, 8);
      lrun[j] = lrun[j] * alpha + rs;
      mrun[j] = mnew;
#pragma unroll
      for (int n = 0; n < 4; ++n) o[n][j] *= alpha;
    }

    // ---- re-layout P: D-fragment -> f16 A-fragment via per-wave LDS ----
#pragma unroll
    for (int n = 0; n < 4; ++n)
#pragma unroll
      for (int j = 0; j < 8; ++j)
        Pw[(j + hi * 8) * LDST + n * 16 + ln] = (_Float16)s[n][j];
    asm volatile("s_wait_dscnt 0" ::: "memory");  // same-wave LDS RAW fence

    // ---- O += P V : 16x64 tile per wave (8 WMMA) ----
    v16h ap0 = load_fragA(Pw, ln, 0, hi);
    v16h ap1 = load_fragA(Pw, ln, 32, hi);
#pragma unroll
    for (int nd = 0; nd < 4; ++nd) {
      v16h bv0 = load_fragB(VTs, nd * 16 + ln, 0, hi);
      v16h bv1 = load_fragB(VTs, nd * 16 + ln, 32, hi);
      o[nd] = __builtin_amdgcn_wmma_f32_16x16x32_f16(false, ap0, false, bv0,
                                                     (short)0, o[nd], false, false);
      o[nd] = __builtin_amdgcn_wmma_f32_16x16x32_f16(false, ap1, false, bv1,
                                                     (short)0, o[nd], false, false);
    }
    __syncthreads();  // all waves done reading K/V/mask LDS before re-staging
  }

  // ---- epilogue: normalize and store f32 output ----
  float* og = out + (size_t)bh * S_LEN * D_DIM;
#pragma unroll
  for (int j = 0; j < 8; ++j) {
    const float inv = 1.0f / lrun[j];
    const int row = q0 + wave * 16 + j + hi * 8;
#pragma unroll
    for (int nd = 0; nd < 4; ++nd)
      og[(size_t)row * D_DIM + nd * 16 + ln] = o[nd][j] * inv;
  }
}

extern "C" void kernel_launch(void* const* d_in, const int* in_sizes, int n_in,
                              void* d_out, int out_size, void* d_ws, size_t ws_size,
                              hipStream_t stream) {
  (void)in_sizes; (void)n_in; (void)d_ws; (void)ws_size; (void)out_size;
  const float* q    = (const float*)d_in[0];
  const float* k    = (const float*)d_in[1];
  const float* v    = (const float*)d_in[2];
  const float* mask = (const float*)d_in[3];
  float* out = (float*)d_out;

  dim3 grid(S_LEN / QT, 4 * 16);  // (q-tiles, B*H)
  dim3 block(128);                // 4 waves of 32
  fa_fwd_kernel<<<grid, block, 0, stream>>>(q, k, v, mask, out);
}
